// ProbSparseSelfAttention_13769665151378
// MI455X (gfx1250) — matrix operand order repaired and checked by reading, not compiled, not run on hardware
//
#include <hip/hip_runtime.h>
#include <hip/hip_bf16.h>

#define DEV static __device__ __forceinline__

typedef __attribute__((ext_vector_type(16))) __bf16 v16bf;
typedef __attribute__((ext_vector_type(8)))  __bf16 v8bf;
typedef __attribute__((ext_vector_type(8)))  float  v8f;
typedef __attribute__((ext_vector_type(4)))  float  v4f;

// ---------- bf16 helpers: use native conversions (backend emits packed cvt) ----------
DEV __bf16 f2bf(float f) { return (__bf16)f; }
DEV float  bf2f(__bf16 b) { return (float)b; }

DEV v8f wmma_bf(v16bf a, v16bf b, v8f c) {
  return __builtin_amdgcn_wmma_f32_16x16x32_bf16(false, a, false, b, (short)0, c, false, false);
}
DEV v16bf cat8(v8bf lo, v8bf hi) {
  return __builtin_shufflevector(lo, hi, 0,1,2,3,4,5,6,7,8,9,10,11,12,13,14,15);
}
DEV v8bf ld8(const __bf16* p) { return *(const v8bf*)p; }

// A-tile (16x32 bf16) from fp32 row: elements 0..7 = K[kb16..kb16+8), 8..15 = K[kb16+8..kb16+16)
DEV v16bf cvtA(const float* p0, const float* p1) {
  v4f x0 = *(const v4f*)p0, x1 = *(const v4f*)(p0 + 4);
  v4f x2 = *(const v4f*)p1, x3 = *(const v4f*)(p1 + 4);
  v16bf r;
#pragma unroll
  for (int i = 0; i < 4; i++) {
    r[i]      = f2bf(x0[i]);
    r[4 + i]  = f2bf(x1[i]);
    r[8 + i]  = f2bf(x2[i]);
    r[12 + i] = f2bf(x3[i]);
  }
  return r;
}

// problem constants
#define BB 4
#define LL 2048
#define DD 1024
#define HH 16
#define DK 64
#define UU 1024
#define MM (BB * LL)   // 8192

// ---------- weight transpose + bf16 convert: WT[n][k] = W[k][n] ----------
__global__ void k_wt(const float* __restrict__ w, __bf16* __restrict__ wt) {
  int idx = blockIdx.x * 256 + threadIdx.x;      // 0 .. 1M-1
  int n = idx >> 10, k = idx & 1023;
  wt[idx] = f2bf(w[k * 1024 + n]);
}

// ---------- WMMA GEMM: out = A(fp32 MxK) @ WT^T + bias ----------
// mode 0: store bf16 into head layout [B,H,L,64]; mode 1: store fp32 row-major (d_out)
__global__ void k_gemm(const float* __restrict__ A, const __bf16* __restrict__ WT,
                       const float* __restrict__ bias, void* __restrict__ out, int mode) {
  const int K = DD;
  int lane = threadIdx.x & 31;
  int wave = threadIdx.x >> 5;
  int m0 = blockIdx.x * 128 + wave * 16;
  int n0 = blockIdx.y * 64;
  int l16  = lane & 15;
  int half = lane >> 4;
  int off  = half * 8;     // unused; A loads use kb16
  int kb16 = half * 16;    // B lane K base

  v8f acc[4] = {{}, {}, {}, {}};
  const float* arow = A + (size_t)(m0 + l16) * K;

  for (int k0 = 0; k0 < K; k0 += 32) {
    v16bf a = cvtA(arow + k0 + kb16, arow + k0 + kb16 + 8);
#pragma unroll
    for (int nt = 0; nt < 4; nt++) {
      const __bf16* bp = WT + (size_t)(n0 + nt * 16 + l16) * K + k0 + kb16;
      v16bf b = *(const v16bf*)bp;
      acc[nt] = wmma_bf(a, b, acc[nt]);
    }
  }

  if (mode == 0) {
    __bf16* O = (__bf16*)out;
#pragma unroll
    for (int nt = 0; nt < 4; nt++) {
      int n = n0 + nt * 16 + l16;
      float bv = bias[n];
      int h = n >> 6, d = n & 63;
#pragma unroll
      for (int r = 0; r < 8; r++) {
        int m = m0 + r + 8 * half;
        int b = m >> 11, lpos = m & 2047;
        size_t o = ((size_t)((b * HH + h) * LL + lpos)) * DK + d;
        O[o] = f2bf(acc[nt][r] + bv);
      }
    }
  } else {
    float* O = (float*)out;
#pragma unroll
    for (int nt = 0; nt < 4; nt++) {
      int n = n0 + nt * 16 + l16;
      float bv = bias[n];
#pragma unroll
      for (int r = 0; r < 8; r++) {
        int m = m0 + r + 8 * half;
        O[(size_t)m * DD + n] = acc[nt][r] + bv;
      }
    }
  }
}

// ---------- top-U by q-norm, stable descending rank ----------
__global__ void k_topk(const __bf16* __restrict__ Qh, int* __restrict__ sel) {
  __shared__ float ns[LL];
  int bh = blockIdx.x;
  for (int i = threadIdx.x; i < LL; i += 256) {
    const __bf16* r = Qh + ((size_t)bh * LL + i) * DK;
    float s = 0.f;
    for (int d = 0; d < DK; d++) { float v = bf2f(r[d]); s += v * v; }
    ns[i] = s;
  }
  __syncthreads();
  for (int i = threadIdx.x; i < LL; i += 256) {
    float vi = ns[i];
    int rank = 0;
    for (int j = 0; j < LL; j++) {
      float vj = ns[j];
      rank += (vj > vi) || (vj == vi && j < i);
    }
    if (rank < UU) sel[bh * UU + rank] = i;
  }
}

// ---------- gather selected rows; pre-scale Q by 1/sqrt(dk); transpose V ----------
__global__ void k_gather(const __bf16* __restrict__ Qh, const __bf16* __restrict__ Kh,
                         const __bf16* __restrict__ Vh, const int* __restrict__ sel,
                         __bf16* __restrict__ Qs, __bf16* __restrict__ Ks,
                         __bf16* __restrict__ Vt) {
  int bh = blockIdx.x;
  int u = blockIdx.y * 64 + (threadIdx.x >> 2);
  int part = (threadIdx.x & 3) * 16;
  int row = sel[bh * UU + u];
  const __bf16* q = Qh + ((size_t)bh * LL + row) * DK + part;
  const __bf16* k = Kh + ((size_t)bh * LL + row) * DK + part;
  const __bf16* v = Vh + ((size_t)bh * LL + row) * DK + part;
  __bf16* qo = Qs + ((size_t)bh * UU + u) * DK + part;
  __bf16* ko = Ks + ((size_t)bh * UU + u) * DK + part;
#pragma unroll
  for (int j = 0; j < 16; j++) {
    qo[j] = f2bf(bf2f(q[j]) * 0.125f);   // 1/sqrt(64)
    ko[j] = k[j];
    Vt[((size_t)bh * DK + part + j) * UU + u] = v[j];
  }
}

__global__ void k_zero(float* __restrict__ p, size_t n) {
  size_t i = (size_t)blockIdx.x * blockDim.x + threadIdx.x;
  if (i < n) p[i] = 0.f;
}

// ---------- flash attention over selected set; S^T trick (keys as M, queries as N) ----------
__global__ void k_attn(const __bf16* __restrict__ Qs, const __bf16* __restrict__ Ks,
                       const __bf16* __restrict__ Vt, const int* __restrict__ sel,
                       float* __restrict__ Ctx) {
  int bh = blockIdx.x;
  int lane = threadIdx.x & 31;
  int wave = threadIdx.x >> 5;
  int q0 = blockIdx.y * 128 + wave * 16;
  int half = lane >> 4;
  int l16 = lane & 15;
  int off8 = half * 8;

  // B-matrices (queries): lane l16 = query column, K = d-chunk
  const __bf16* qbase = Qs + ((size_t)bh * UU + q0 + l16) * DK;
  v16bf bq0 = cat8(ld8(qbase + off8), ld8(qbase + off8 + 16));             // d = 0..31
  v16bf bq1 = cat8(ld8(qbase + 32 + off8), ld8(qbase + 32 + off8 + 16));   // d = 32..63

  float m_i = -__builtin_inff();
  float l_i = 0.f;
  v8f O0 = {}, O1 = {}, O2 = {}, O3 = {};

  for (int w0 = 0; w0 < UU; w0 += 32) {
    // A-matrices: rows = keys
    const __bf16* ka = Ks + ((size_t)bh * UU + w0 + l16) * DK;
    const __bf16* kb = Ks + ((size_t)bh * UU + w0 + 16 + l16) * DK;
    v16bf Aa0 = cat8(ld8(ka + off8),      ld8(ka + off8 + 16));
    v16bf Aa1 = cat8(ld8(ka + 32 + off8), ld8(ka + 32 + off8 + 16));
    v16bf Ab0 = cat8(ld8(kb + off8),      ld8(kb + off8 + 16));
    v16bf Ab1 = cat8(ld8(kb + 32 + off8), ld8(kb + 32 + off8 + 16));

    v8f z = {};
    v8f Sa = wmma_bf(Aa0, bq0, z); Sa = wmma_bf(Aa1, bq1, Sa);   // keys w0..w0+15
    v8f Sb = wmma_bf(Ab0, bq0, z); Sb = wmma_bf(Ab1, bq1, Sb);   // keys w0+16..w0+31

    // online softmax: lane owns query l16; Sa/Sb rows are keys
    float mx = -__builtin_inff();
#pragma unroll
    for (int r = 0; r < 8; r++) { mx = fmaxf(mx, Sa[r]); mx = fmaxf(mx, Sb[r]); }
    mx = fmaxf(mx, __shfl_xor(mx, 16, 32));
    float m_new = fmaxf(m_i, mx);
    float sc = __expf(m_i - m_new);

    float Pa[8], Pb[8], rs = 0.f;
#pragma unroll
    for (int r = 0; r < 8; r++) {
      Pa[r] = __expf(Sa[r] - m_new);
      Pb[r] = __expf(Sb[r] - m_new);
      rs += Pa[r] + Pb[r];
    }
    rs += __shfl_xor(rs, 16, 32);
    l_i = l_i * sc + rs;
    m_i = m_new;

    // pack P into A layout for P@V: C layout of S^T already matches per-lane
    v16bf PA;
#pragma unroll
    for (int r = 0; r < 8; r++) { PA[r] = f2bf(Pa[r]); PA[8 + r] = f2bf(Pb[r]); }

    // rescale accumulators per query-row (row = r + 8*half)
#pragma unroll
    for (int r = 0; r < 8; r++) {
      float s_r = __shfl(sc, r + 8 * half, 32);
      O0[r] *= s_r; O1[r] *= s_r; O2[r] *= s_r; O3[r] *= s_r;
    }

    // V B-tiles from transposed V: lane = d column, K = keys
    const __bf16* vb = Vt + (size_t)bh * DK * UU;
    v16bf BV0 = cat8(ld8(vb + (size_t)(0  + l16) * UU + w0 + off8), ld8(vb + (size_t)(0  + l16) * UU + w0 + off8 + 16));
    v16bf BV1 = cat8(ld8(vb + (size_t)(16 + l16) * UU + w0 + off8), ld8(vb + (size_t)(16 + l16) * UU + w0 + off8 + 16));
    v16bf BV2 = cat8(ld8(vb + (size_t)(32 + l16) * UU + w0 + off8), ld8(vb + (size_t)(32 + l16) * UU + w0 + off8 + 16));
    v16bf BV3 = cat8(ld8(vb + (size_t)(48 + l16) * UU + w0 + off8), ld8(vb + (size_t)(48 + l16) * UU + w0 + off8 + 16));
    O0 = wmma_bf(PA, BV0, O0);
    O1 = wmma_bf(PA, BV1, O1);
    O2 = wmma_bf(PA, BV2, O2);
    O3 = wmma_bf(PA, BV3, O3);
  }

  // normalize and scatter back to [B, L, H*64 + d]
  int b = bh >> 4, h = bh & 15;
#pragma unroll
  for (int r = 0; r < 8; r++) {
    float lr = __shfl(l_i, r + 8 * half, 32);
    float inv = 1.0f / lr;
    int row = sel[bh * UU + q0 + r + 8 * half];
    size_t base = ((size_t)(b * LL + row)) * DD + h * DK + l16;
    Ctx[base + 0]  = O0[r] * inv;
    Ctx[base + 16] = O1[r] * inv;
    Ctx[base + 32] = O2[r] * inv;
    Ctx[base + 48] = O3[r] * inv;
  }
}

extern "C" void kernel_launch(void* const* d_in, const int* in_sizes, int n_in,
                              void* d_out, int out_size, void* d_ws, size_t ws_size,
                              hipStream_t stream) {
  (void)in_sizes; (void)n_in; (void)out_size; (void)ws_size;
  const size_t MB = 1u << 20;
  char* w = (char*)d_ws;
  __bf16* wqT = (__bf16*)(w + 0 * MB);
  __bf16* wkT = (__bf16*)(w + 2 * MB);
  __bf16* wvT = (__bf16*)(w + 4 * MB);
  __bf16* woT = (__bf16*)(w + 6 * MB);
  __bf16* Qh  = (__bf16*)(w + 8 * MB);    // [B,H,L,64] bf16  16MB
  __bf16* Kh  = (__bf16*)(w + 24 * MB);
  __bf16* Vh  = (__bf16*)(w + 40 * MB);
  int*    sel = (int*)   (w + 56 * MB);   // [BH,U]           256KB
  __bf16* Qs  = (__bf16*)(w + 57 * MB);   // [BH,U,64]        8MB
  __bf16* Ks  = (__bf16*)(w + 65 * MB);
  __bf16* Vt  = (__bf16*)(w + 73 * MB);   // [BH,64,U]        8MB
  float*  Ctx = (float*) (w + 81 * MB);   // [B,L,D] fp32     32MB

  const float* queries = (const float*)d_in[0];
  const float* keys    = (const float*)d_in[1];
  const float* values  = (const float*)d_in[2];

  dim3 b256(256);
  k_wt<<<4096, b256, 0, stream>>>((const float*)d_in[3], wqT);
  k_wt<<<4096, b256, 0, stream>>>((const float*)d_in[5], wkT);
  k_wt<<<4096, b256, 0, stream>>>((const float*)d_in[7], wvT);
  k_wt<<<4096, b256, 0, stream>>>((const float*)d_in[9], woT);

  dim3 gg(MM / 128, DD / 64);   // (64, 16)
  k_gemm<<<gg, b256, 0, stream>>>(queries, wqT, (const float*)d_in[4], Qh, 0);
  k_gemm<<<gg, b256, 0, stream>>>(keys,    wkT, (const float*)d_in[6], Kh, 0);
  k_gemm<<<gg, b256, 0, stream>>>(values,  wvT, (const float*)d_in[8], Vh, 0);

  k_topk<<<BB * HH, b256, 0, stream>>>(Qh, sel);
  k_gather<<<dim3(BB * HH, UU / 64), b256, 0, stream>>>(Qh, Kh, Vh, sel, Qs, Ks, Vt);
  k_zero<<<(unsigned)((size_t)MM * DD / 256), b256, 0, stream>>>(Ctx, (size_t)MM * DD);
  k_attn<<<dim3(BB * HH, UU / 128), b256, 0, stream>>>(Qs, Ks, Vt, sel, Ctx);

  k_gemm<<<gg, b256, 0, stream>>>(Ctx, woT, (const float*)d_in[10], d_out, 1);
}
